// GPT_61383672594593
// MI455X (gfx1250) — matrix operand order repaired and checked by reading, not compiled, not run on hardware
//
#include <hip/hip_runtime.h>

// ---------------------------------------------------------------------------
// GPT-2-small-ish forward (L=4, B=2, S=1024, D=768, H=12, FF=3072, V=50257)
// for MI455X (gfx1250): all matmuls via v_wmma_f32_16x16x32_bf16 (wave32),
// fed by double-buffered LDS tiles filled with global_load_async_to_lds_b128
// (ASYNCcnt) — the CDNA5 async data-movement path.
// ---------------------------------------------------------------------------

typedef __bf16 bf16;
typedef bf16  v16bf __attribute__((ext_vector_type(16)));
typedef float v8f   __attribute__((ext_vector_type(8)));

union Frag16 { v16bf v; uint4 q[2]; unsigned u[8]; };

constexpr int Lc = 4, Bc = 2, Sc = 1024, Dc = 768, Hc = 12, FFc = 3072, Vc = 50257;
constexpr int TD = 3 * Dc;        // 2304
constexpr int HD = 64;            // head dim
constexpr int Mrows = Bc * Sc;    // 2048

enum { FLAG_BIAS = 1, FLAG_GELU = 2, FLAG_RESID = 4, FLAG_CAUSAL = 8 };

__device__ __forceinline__ unsigned short f2bf(float f) {
  unsigned u = __float_as_uint(f);
  u += 0x7FFFu + ((u >> 16) & 1u);      // round-to-nearest-even
  return (unsigned short)(u >> 16);
}

// One 16-byte async DMA chunk: global -> LDS, tracked by ASYNCcnt.
__device__ __forceinline__ void async_copy16(unsigned ldsOff, const void* g) {
  asm volatile("global_load_async_to_lds_b128 %0, %1, off"
               :: "v"(ldsOff), "v"((unsigned long long)(size_t)g)
               : "memory");
}

__device__ __forceinline__ void wait_async0() {
  asm volatile("s_wait_asynccnt 0x0" ::: "memory");
}

// ---------------------------------------------------------------------------
// Generic bf16 WMMA GEMM:  C[M,N](f32) = A[M,K](bf16) * W[N,K](bf16)^T
// Batched over blockIdx.z = b*Hdec + h with independent (b,h) strides.
// Block = 256 threads = 8 waves (2 along M x 4 along N); wave tile 64x32;
// block tile 128x128, K-step 32, double-buffered async LDS staging.
// All K multiples of 32; all M multiples of 128; N bounds-handled
// (N=64 AV GEMM, N=50257 lm_head) by clamping B rows + guarded stores.
// ---------------------------------------------------------------------------
__global__ __launch_bounds__(256) void gemm_bf16_wmma(
    const unsigned short* __restrict__ A, long long sAb, long long sAh, int lda,
    const unsigned short* __restrict__ Bw, long long sBb, long long sBh, int ldb,
    float* __restrict__ C, long long sCb, long long sCh, int ldc,
    const float* __restrict__ bias, const float* __restrict__ resid, int ldr,
    int M, int N, int K, int Hdec, float scale, int flags)
{
  __shared__ unsigned short ldsA[2][128 * 32];   // 8 KB per buffer
  __shared__ unsigned short ldsB[2][128 * 32];   // 8 KB per buffer

  const int z  = blockIdx.z;
  const int zb = z / Hdec, zh = z % Hdec;
  A  += zb * sAb + zh * sAh;
  Bw += zb * sBb + zh * sBh;
  C  += zb * sCb + zh * sCh;

  const int tid  = threadIdx.x;
  const int lane = tid & 31;
  const int wave = tid >> 5;
  const int wm   = wave & 1;            // 2 waves along M
  const int wn   = wave >> 1;           // 4 waves along N
  const int mBlk = blockIdx.y * 128;
  const int nBlk = blockIdx.x * 128;
  const int half = lane >> 4;           // 0/1
  const int lr   = lane & 15;

  // LDS byte offsets of the two buffers (low 32 bits of flat shared address).
  const unsigned aOff0 = (unsigned)(size_t)&ldsA[0][0];
  const unsigned aOff1 = (unsigned)(size_t)&ldsA[1][0];
  const unsigned bOff0 = (unsigned)(size_t)&ldsB[0][0];
  const unsigned bOff1 = (unsigned)(size_t)&ldsB[1][0];

  // Cooperative async fill of one K-step tile pair into buffer `buf`.
  // Tile = 128 rows x 32 bf16 = 512 chunks of 16B; 256 threads x 2 chunks.
  // 16B chunks are XOR-swizzled: phys = chunk ^ ((row>>2)&3)  -> conflict-free
  // ds reads (16 consecutive rows x 16B per fragment load).
  auto issue = [&](int k0, int buf) {
    const unsigned aBase = buf ? aOff1 : aOff0;
    const unsigned bBase = buf ? bOff1 : bOff0;
#pragma unroll
    for (int t = 0; t < 2; ++t) {
      const int c   = tid + t * 256;        // 0..511
      const int row = c >> 2;               // 0..127
      const int ch  = c & 3;                // 16B chunk within 64B row
      const int sw  = (row >> 2) & 3;
      const unsigned dOff = (unsigned)(row * 64 + ((ch ^ sw) << 4));
      async_copy16(aBase + dOff, A + (size_t)(mBlk + row) * lda + k0 + ch * 8);
      int bn = nBlk + row; if (bn >= N) bn = N - 1;   // clamp ragged N
      async_copy16(bBase + dOff, Bw + (size_t)bn * ldb + k0 + ch * 8);
    }
  };

  v8f acc[4][2] = {};
  const int nk = K >> 5;
  issue(0, 0);

  for (int s = 0; s < nk; ++s) {
    wait_async0();        // my DMA chunks for buffer s&1 have landed
    __syncthreads();      // everyone's landed; buffer (s+1)&1 free to overwrite
    if (s + 1 < nk) issue((s + 1) << 5, (s + 1) & 1);

    const int buf = s & 1;
    // A fragments (16x32): lane = row lr, logical 16B chunks {half, half+2}.
    Frag16 af[4];
#pragma unroll
    for (int i = 0; i < 4; ++i) {
      const int row = wm * 64 + i * 16 + lr;
      const int sw  = (row >> 2) & 3;
      const uint4* rp = (const uint4*)&ldsA[buf][row * 32];
      af[i].q[0] = rp[half ^ sw];
      af[i].q[1] = rp[(half + 2) ^ sw];
    }
    // B fragments (32x16): lane = col lr, logical chunks {half, half+2}.
    Frag16 bfr[2];
#pragma unroll
    for (int j = 0; j < 2; ++j) {
      const int row = wn * 32 + j * 16 + lr;
      const int sw  = (row >> 2) & 3;
      const uint4* rp = (const uint4*)&ldsB[buf][row * 32];
      bfr[j].q[0] = rp[half ^ sw];
      bfr[j].q[1] = rp[(half + 2) ^ sw];
    }
#pragma unroll
    for (int i = 0; i < 4; ++i)
#pragma unroll
      for (int j = 0; j < 2; ++j)
        acc[i][j] = __builtin_amdgcn_wmma_f32_16x16x32_bf16(
            false, af[i].v, false, bfr[j].v, (short)0, acc[i][j], false, false);
  }

  // Epilogue: C/D layout -> lane holds n = lr, word v holds m = 8*half + v.
#pragma unroll
  for (int i = 0; i < 4; ++i) {
#pragma unroll
    for (int j = 0; j < 2; ++j) {
      const int n = nBlk + wn * 32 + j * 16 + lr;
      if (n >= N) continue;
#pragma unroll
      for (int v = 0; v < 8; ++v) {
        const int m = mBlk + wm * 64 + i * 16 + half * 8 + v;
        float val = acc[i][j][v] * scale;
        if (flags & FLAG_CAUSAL) { if (n > m) val = -1e30f; }
        if (flags & FLAG_BIAS)   val += bias[n];
        if (flags & FLAG_GELU)   val = 0.5f * val * (1.0f + erff(val * 0.70710678118654752f));
        if (flags & FLAG_RESID)  val += resid[(size_t)m * ldr + n];
        C[(size_t)m * ldc + n] = val;
      }
    }
  }
}

// ---------------------------------------------------------------------------
// LayerNorm over D=768, one block per row, bf16 output (feeds WMMA A operand).
// ---------------------------------------------------------------------------
__global__ __launch_bounds__(256) void layernorm_to_bf16(
    const float* __restrict__ x, const float* __restrict__ w,
    const float* __restrict__ b, unsigned short* __restrict__ out, int Dm)
{
  __shared__ float red[256];
  const int row = blockIdx.x, tid = threadIdx.x;
  const float* xr = x + (size_t)row * Dm;

  float s = 0.f;
  for (int d = tid; d < Dm; d += 256) s += xr[d];
  red[tid] = s; __syncthreads();
  for (int st = 128; st > 0; st >>= 1) { if (tid < st) red[tid] += red[tid + st]; __syncthreads(); }
  const float mean = red[0] / Dm;
  __syncthreads();

  float v = 0.f;
  for (int d = tid; d < Dm; d += 256) { float t = xr[d] - mean; v += t * t; }
  red[tid] = v; __syncthreads();
  for (int st = 128; st > 0; st >>= 1) { if (tid < st) red[tid] += red[tid + st]; __syncthreads(); }
  const float rstd = rsqrtf(red[0] / Dm + 1e-5f);

  for (int d = tid; d < Dm; d += 256)
    out[(size_t)row * Dm + d] = f2bf((xr[d] - mean) * rstd * w[d] + b[d]);
}

// ---------------------------------------------------------------------------
// Row softmax over S=1024 (masked entries hold -1e30 -> exp underflows to 0),
// bf16 probabilities out (feeds WMMA A operand of the P*V GEMM).
// ---------------------------------------------------------------------------
__global__ __launch_bounds__(256) void softmax_bf16(
    const float* __restrict__ sc, unsigned short* __restrict__ pr)
{
  __shared__ float red[256];
  const int row = blockIdx.x, tid = threadIdx.x;
  const float* s = sc + (size_t)row * Sc;
  unsigned short* p = pr + (size_t)row * Sc;

  float mx = -3.4e38f;
  for (int j = tid; j < Sc; j += 256) mx = fmaxf(mx, s[j]);
  red[tid] = mx; __syncthreads();
  for (int st = 128; st > 0; st >>= 1) { if (tid < st) red[tid] = fmaxf(red[tid], red[tid + st]); __syncthreads(); }
  mx = red[0]; __syncthreads();

  float sum = 0.f;
  for (int j = tid; j < Sc; j += 256) sum += expf(s[j] - mx);
  red[tid] = sum; __syncthreads();
  for (int st = 128; st > 0; st >>= 1) { if (tid < st) red[tid] += red[tid + st]; __syncthreads(); }
  const float inv = 1.0f / red[0];

  for (int j = tid; j < Sc; j += 256) p[j] = f2bf(expf(s[j] - mx) * inv);
}

// x[b,s,:] = tok_emb[id] + pos_emb[s]
__global__ void embed_kernel(const int* __restrict__ ids, const float* __restrict__ tok,
                             const float* __restrict__ pos, float* __restrict__ x)
{
  size_t i = (size_t)blockIdx.x * blockDim.x + threadIdx.x;
  if (i >= (size_t)Mrows * Dc) return;
  const int m = (int)(i / Dc), d = (int)(i % Dc), s = m % Sc;
  x[i] = tok[(size_t)ids[m] * Dc + d] + pos[(size_t)s * Dc + d];
}

__global__ void cvt_f32_bf16(const float* __restrict__ in, unsigned short* __restrict__ out, size_t n)
{
  size_t i = (size_t)blockIdx.x * blockDim.x + threadIdx.x;
  const size_t stride = (size_t)gridDim.x * blockDim.x;
  for (; i < n; i += stride) out[i] = f2bf(in[i]);
}

// vt[(b*H+h)][d][k] = (bf16) qkv_f32[b*S+k][2D + h*64 + d]   (V^T per head)
__global__ void vtranspose_kernel(const float* __restrict__ qkv, unsigned short* __restrict__ vt)
{
  size_t i = (size_t)blockIdx.x * blockDim.x + threadIdx.x;
  if (i >= (size_t)Bc * Hc * HD * Sc) return;
  const int k = (int)(i % Sc);
  const int d = (int)((i / Sc) % HD);
  const int z = (int)(i / ((size_t)Sc * HD));
  const int b = z / Hc, h = z % Hc;
  vt[i] = f2bf(qkv[((size_t)(b * Sc + k)) * TD + 2 * Dc + h * HD + d]);
}

// ---------------------------------------------------------------------------
extern "C" void kernel_launch(void* const* d_in, const int* in_sizes, int n_in,
                              void* d_out, int out_size, void* d_ws, size_t ws_size,
                              hipStream_t stream)
{
  (void)in_sizes; (void)n_in; (void)out_size; (void)ws_size;
  const int*   ids  = (const int*)  d_in[0];
  const float* tok  = (const float*)d_in[1];
  const float* pos  = (const float*)d_in[2];
  const float* ln1w = (const float*)d_in[3];
  const float* ln1b = (const float*)d_in[4];
  const float* qkvw = (const float*)d_in[5];
  const float* qkvb = (const float*)d_in[6];
  const float* outw = (const float*)d_in[7];
  const float* outb = (const float*)d_in[8];
  const float* ln2w = (const float*)d_in[9];
  const float* ln2b = (const float*)d_in[10];
  const float* fc1w = (const float*)d_in[11];
  const float* fc1b = (const float*)d_in[12];
  const float* fc2w = (const float*)d_in[13];
  const float* fc2b = (const float*)d_in[14];
  const float* lnfw = (const float*)d_in[15];
  const float* lnfb = (const float*)d_in[16];
  float* logits = (float*)d_out;

  // workspace carve-out (~370 MB)
  char* wp = (char*)d_ws;
  auto carve = [&](size_t bytes) { char* p = wp; wp += (bytes + 255) & ~(size_t)255; return p; };
  unsigned short* tokbf  = (unsigned short*)carve((size_t)Vc * Dc * 2);
  unsigned short* qkvwbf = (unsigned short*)carve((size_t)Lc * TD * Dc * 2);
  unsigned short* outwbf = (unsigned short*)carve((size_t)Lc * Dc * Dc * 2);
  unsigned short* fc1wbf = (unsigned short*)carve((size_t)Lc * FFc * Dc * 2);
  unsigned short* fc2wbf = (unsigned short*)carve((size_t)Lc * Dc * FFc * 2);
  float*          x      = (float*)         carve((size_t)Mrows * Dc * 4);
  unsigned short* hbf    = (unsigned short*)carve((size_t)Mrows * FFc * 2);
  float*          qkvf   = (float*)         carve((size_t)Mrows * TD * 4);
  unsigned short* qkvbf  = (unsigned short*)carve((size_t)Mrows * TD * 2);
  float*          scores = (float*)         carve((size_t)Bc * Hc * Sc * Sc * 4);
  unsigned short* probs  = (unsigned short*)carve((size_t)Bc * Hc * Sc * Sc * 2);
  unsigned short* vt     = (unsigned short*)carve((size_t)Bc * Hc * HD * Sc * 2);
  float*          abuf   = (float*)         carve((size_t)Mrows * Dc * 4);
  float*          hff    = (float*)         carve((size_t)Mrows * FFc * 4);

  // weights -> bf16 (every call; deterministic)
  cvt_f32_bf16<<<4096, 256, 0, stream>>>(tok,  tokbf,  (size_t)Vc * Dc);
  cvt_f32_bf16<<<4096, 256, 0, stream>>>(qkvw, qkvwbf, (size_t)Lc * TD * Dc);
  cvt_f32_bf16<<<4096, 256, 0, stream>>>(outw, outwbf, (size_t)Lc * Dc * Dc);
  cvt_f32_bf16<<<4096, 256, 0, stream>>>(fc1w, fc1wbf, (size_t)Lc * FFc * Dc);
  cvt_f32_bf16<<<4096, 256, 0, stream>>>(fc2w, fc2wbf, (size_t)Lc * Dc * FFc);

  embed_kernel<<<(Mrows * Dc + 255) / 256, 256, 0, stream>>>(ids, tok, pos, x);

  for (int l = 0; l < Lc; ++l) {
    // --- attention ---
    layernorm_to_bf16<<<Mrows, 256, 0, stream>>>(x, ln1w + l * Dc, ln1b + l * Dc, hbf, Dc);
    gemm_bf16_wmma<<<dim3(TD / 128, Mrows / 128, 1), 256, 0, stream>>>(
        hbf, 0, 0, Dc, qkvwbf + (size_t)l * TD * Dc, 0, 0, Dc,
        qkvf, 0, 0, TD, qkvb + (size_t)l * TD, nullptr, 0,
        Mrows, TD, Dc, 1, 1.0f, FLAG_BIAS);
    cvt_f32_bf16<<<4096, 256, 0, stream>>>(qkvf, qkvbf, (size_t)Mrows * TD);
    vtranspose_kernel<<<((size_t)Bc * Hc * HD * Sc + 255) / 256, 256, 0, stream>>>(qkvf, vt);

    // scores[b,h] = Q K^T / 8 with causal mask; batch over z=(b,h)
    gemm_bf16_wmma<<<dim3(Sc / 128, Sc / 128, Bc * Hc), 256, 0, stream>>>(
        qkvbf,      (long long)Sc * TD, 64, TD,
        qkvbf + Dc, (long long)Sc * TD, 64, TD,
        scores, (long long)Hc * Sc * Sc, (long long)Sc * Sc, Sc,
        nullptr, nullptr, 0,
        Sc, Sc, HD, Hc, 0.125f, FLAG_CAUSAL);
    softmax_bf16<<<Bc * Hc * Sc, 256, 0, stream>>>(scores, probs);

    // attn[b,h] = P V  (V^T is row-major [HD, S] per head)
    gemm_bf16_wmma<<<dim3(1, Sc / 128, Bc * Hc), 256, 0, stream>>>(
        probs, (long long)Hc * Sc * Sc, (long long)Sc * Sc, Sc,
        vt,    (long long)Hc * HD * Sc, (long long)HD * Sc, Sc,
        abuf,  (long long)Sc * Dc, HD, Dc,
        nullptr, nullptr, 0,
        Sc, HD, Sc, Hc, 1.0f, 0);
    cvt_f32_bf16<<<4096, 256, 0, stream>>>(abuf, hbf, (size_t)Mrows * Dc);

    // x += attn @ out_w^T + out_b
    gemm_bf16_wmma<<<dim3(Dc / 128, Mrows / 128, 1), 256, 0, stream>>>(
        hbf, 0, 0, Dc, outwbf + (size_t)l * Dc * Dc, 0, 0, Dc,
        x, 0, 0, Dc, outb + (size_t)l * Dc, x, Dc,
        Mrows, Dc, Dc, 1, 1.0f, FLAG_BIAS | FLAG_RESID);

    // --- ffn ---
    layernorm_to_bf16<<<Mrows, 256, 0, stream>>>(x, ln2w + l * Dc, ln2b + l * Dc, hbf, Dc);
    gemm_bf16_wmma<<<dim3(FFc / 128, Mrows / 128, 1), 256, 0, stream>>>(
        hbf, 0, 0, Dc, fc1wbf + (size_t)l * FFc * Dc, 0, 0, Dc,
        hff, 0, 0, FFc, fc1b + (size_t)l * FFc, nullptr, 0,
        Mrows, FFc, Dc, 1, 1.0f, FLAG_BIAS | FLAG_GELU);
    cvt_f32_bf16<<<4096, 256, 0, stream>>>(hff, hbf, (size_t)Mrows * FFc);
    gemm_bf16_wmma<<<dim3(Dc / 128, Mrows / 128, 1), 256, 0, stream>>>(
        hbf, 0, 0, FFc, fc2wbf + (size_t)l * Dc * FFc, 0, 0, FFc,
        x, 0, 0, Dc, fc2b + (size_t)l * Dc, x, Dc,
        Mrows, Dc, FFc, 1, 1.0f, FLAG_BIAS | FLAG_RESID);
  }

  // final LN + tied lm_head (dominant GEMM: 2048 x 50257 x 768)
  layernorm_to_bf16<<<Mrows, 256, 0, stream>>>(x, lnfw, lnfb, hbf, Dc);
  gemm_bf16_wmma<<<dim3((Vc + 127) / 128, Mrows / 128, 1), 256, 0, stream>>>(
      hbf, 0, 0, Dc, tokbf, 0, 0, Dc,
      logits, 0, 0, Vc, nullptr, nullptr, 0,
      Mrows, Vc, Dc, 1, 1.0f, 0);
}
